// SBLocalAttention_71674414235764
// MI455X (gfx1250) — matrix operand order repaired and checked by reading, not compiled, not run on hardware
//
#include <hip/hip_runtime.h>
#include <math.h>

#define BB 2
#define LL 1024
#define DM 1024
#define HH 16
#define DH 64
#define MF 24
#define MP 32
#define WIN 128
#define NEG_MASK (-10000.0f)
#define FEPS 1e-4f
#define SCL 0.125f                       /* 1/sqrt(DH) */
#define DATA_NORM 0.35355339059327373f   /* sqrt(SCL)  */

typedef __attribute__((ext_vector_type(16))) __bf16 v16bf;
typedef __attribute__((ext_vector_type(8)))  float  v8f;
typedef __attribute__((ext_vector_type(4)))  unsigned int u32x4;

union FragU { u32x4 q[2]; unsigned short u[16]; v16bf v; };

__device__ __forceinline__ unsigned short f2bf(float f) {
  unsigned int u = __float_as_uint(f);
  u += 0x7fffu + ((u >> 16) & 1u);
  return (unsigned short)(u >> 16);
}
__device__ __forceinline__ unsigned int encf(float f) {
  unsigned int u = __float_as_uint(f);
  return (u & 0x80000000u) ? ~u : (u | 0x80000000u);
}
__device__ __forceinline__ float decf(unsigned int u) {
  return __uint_as_float((u & 0x80000000u) ? (u ^ 0x80000000u) : ~u);
}

// A fragment (16x32 bf16): lane holds row (lane&15); K = half*16 .. +15 contiguous,
// 2 elems per VGPR. rowp = &A[(m0 + (lane&15)) * ld].
__device__ __forceinline__ v16bf load_a_row(const unsigned short* rowp, int k0, int half) {
  FragU f;
  const unsigned short* p = rowp + k0 + half * 16;
  f.q[0] = *(const u32x4*)p;
  f.q[1] = *(const u32x4*)(p + 8);
  return f.v;
}
// B fragment (32x16 bf16): lane holds column (lane&15); K = half*16 .. +15 contiguous.
// rowp = &W[(n0 + (lane&15)) * ld]  (W row-major over K, i.e. B[k][n] = W[n][k]).
__device__ __forceinline__ v16bf load_b_row(const unsigned short* rowp, int k0, int half) {
  FragU f;
  const unsigned short* p = rowp + k0 + half * 16;
  f.q[0] = *(const u32x4*)p;
  f.q[1] = *(const u32x4*)(p + 8);
  return f.v;
}

// ---------------------------------------------------------------- conversions
__global__ void sb_cvt_f32_to_bf16(const float* __restrict__ s,
                                   unsigned short* __restrict__ d, int n) {
  for (int i = blockIdx.x * blockDim.x + threadIdx.x; i < n; i += gridDim.x * blockDim.x)
    d[i] = f2bf(s[i]);
}

__global__ void sb_extract_qkv_bf16(const float* __restrict__ qkv,
                                    unsigned short* __restrict__ dst, int sel) {
  const int n = BB * HH * LL * DH;
  for (int i = blockIdx.x * blockDim.x + threadIdx.x; i < n; i += gridDim.x * blockDim.x) {
    int d = i & 63;
    int l = (i >> 6) & (LL - 1);
    int h = (i >> 16) & (HH - 1);
    int b = i >> 20;
    dst[i] = f2bf(qkv[((size_t)(b * LL + l)) * (3 * DM) + sel * DM + h * DH + d]);
  }
}

__global__ void sb_init_kls(unsigned int* k) {
  if ((int)threadIdx.x < BB * HH) k[threadIdx.x] = 0u;
}

// ---------------------------------------------------------------- WMMA GEMM
// C[M,N] = A[M,K](bf16,row-major) * W[N,K](bf16,row-major)^T + bias[N]
// Block tile 128x64, 8 waves, each wave owns a 32x32 tile (2x2 WMMA tiles).
__global__ __launch_bounds__(256) void sb_gemm_bf16_wmma(
    const unsigned short* __restrict__ A, const unsigned short* __restrict__ Bw,
    const float* __restrict__ bias, float* __restrict__ C,
    int Mdim, int Ndim, int K) {
  int wave = threadIdx.x >> 5;
  int lane = threadIdx.x & 31;
  int half = lane >> 4, ln = lane & 15;
  int m0 = blockIdx.x * 128 + (wave & 3) * 32;
  int n0 = blockIdx.y * 64 + (wave >> 2) * 32;
  const unsigned short* arow0 = A  + (size_t)(m0 + ln) * K;
  const unsigned short* arow1 = A  + (size_t)(m0 + 16 + ln) * K;
  const unsigned short* brow0 = Bw + (size_t)(n0 + ln) * K;
  const unsigned short* brow1 = Bw + (size_t)(n0 + 16 + ln) * K;
  v8f c00 = {}; v8f c01 = {}; v8f c10 = {}; v8f c11 = {};
  for (int k0 = 0; k0 < K; k0 += 32) {
    if (k0 + 64 < K) {                       // prefetch next slab -> global_prefetch
      __builtin_prefetch(arow0 + k0 + 64, 0, 3);
      __builtin_prefetch(arow1 + k0 + 64, 0, 3);
      __builtin_prefetch(brow0 + k0 + 64, 0, 3);
      __builtin_prefetch(brow1 + k0 + 64, 0, 3);
    }
    v16bf a0 = load_a_row(arow0, k0, half);
    v16bf a1 = load_a_row(arow1, k0, half);
    v16bf b0 = load_b_row(brow0, k0, half);
    v16bf b1 = load_b_row(brow1, k0, half);
    c00 = __builtin_amdgcn_wmma_f32_16x16x32_bf16(false, a0, false, b0, (short)0, c00, false, false);
    c01 = __builtin_amdgcn_wmma_f32_16x16x32_bf16(false, a0, false, b1, (short)0, c01, false, false);
    c10 = __builtin_amdgcn_wmma_f32_16x16x32_bf16(false, a1, false, b0, (short)0, c10, false, false);
    c11 = __builtin_amdgcn_wmma_f32_16x16x32_bf16(false, a1, false, b1, (short)0, c11, false, false);
  }
  float bv0 = bias[n0 + ln];
  float bv1 = bias[n0 + 16 + ln];
#pragma unroll
  for (int r = 0; r < 8; ++r) {
    int rowa = m0 + r + half * 8;
    int rowb = m0 + 16 + r + half * 8;
    C[(size_t)rowa * Ndim + n0 + ln]      = c00[r] + bv0;
    C[(size_t)rowa * Ndim + n0 + 16 + ln] = c01[r] + bv1;
    C[(size_t)rowb * Ndim + n0 + ln]      = c10[r] + bv0;
    C[(size_t)rowb * Ndim + n0 + 16 + ln] = c11[r] + bv1;
  }
}

// ---------------------------------------------------------------- feature maps
__global__ __launch_bounds__(256) void sb_feat_q(
    const float* __restrict__ qkv, const float* __restrict__ proj,
    float* __restrict__ qp32, unsigned short* __restrict__ qpb,
    float* __restrict__ qls) {
  int idx = blockIdx.x * blockDim.x + threadIdx.x;   // exact grid: BB*HH*LL
  int l = idx & (LL - 1);
  int h = (idx >> 10) & (HH - 1);
  int b = idx >> 14;
  const float* qrow = qkv + ((size_t)(b * LL + l)) * (3 * DM) + h * DH;
  float qr[DH];
  float diag = 0.f;
  for (int d = 0; d < DH; ++d) { float v = qrow[d] * DATA_NORM; qr[d] = v; diag += v * v; }
  diag *= 0.5f;
  float lf[MF];
  float mx = -3e38f;
  for (int m = 0; m < MF; ++m) {
    float dot = 0.f;
    const float* pr = proj + m * DH;
    for (int d = 0; d < DH; ++d) dot += qr[d] * pr[d];
    lf[m] = dot - diag;
    mx = fmaxf(mx, lf[m]);
  }
  qls[idx] = mx;
  for (int m = 0; m < MP; ++m) {
    float p = (m < MF) ? (__expf(lf[m] - mx) + FEPS) : 0.f;
    qp32[(size_t)idx * MP + m] = p;
    qpb[(size_t)idx * MP + m]  = f2bf(p);
  }
}

__global__ __launch_bounds__(256) void sb_feat_k1(
    const float* __restrict__ qkv, const float* __restrict__ proj,
    float* __restrict__ klf, unsigned int* __restrict__ klsbits) {
  __shared__ float redm[256];
  int idx = blockIdx.x * blockDim.x + threadIdx.x;   // exact grid: BB*HH*LL
  int l = idx & (LL - 1);
  int h = (idx >> 10) & (HH - 1);
  int b = idx >> 14;
  const float* krow = qkv + ((size_t)(b * LL + l)) * (3 * DM) + DM + h * DH;
  float kr[DH];
  float diag = 0.f;
  for (int d = 0; d < DH; ++d) { float v = krow[d] * DATA_NORM; kr[d] = v; diag += v * v; }
  diag *= 0.5f;
  float mx = -3e38f;
  for (int m = 0; m < MF; ++m) {
    float dot = 0.f;
    const float* pr = proj + m * DH;
    for (int d = 0; d < DH; ++d) dot += kr[d] * pr[d];
    float lf = dot - diag;
    klf[(size_t)idx * MP + m] = lf;
    mx = fmaxf(mx, lf);
  }
  for (int m = MF; m < MP; ++m) klf[(size_t)idx * MP + m] = 0.f;
  redm[threadIdx.x] = mx;
  __syncthreads();
  for (int s = 128; s > 0; s >>= 1) {
    if ((int)threadIdx.x < s)
      redm[threadIdx.x] = fmaxf(redm[threadIdx.x], redm[threadIdx.x + s]);
    __syncthreads();
  }
  if (threadIdx.x == 0) atomicMax(&klsbits[b * HH + h], encf(redm[0]));
}

__global__ void sb_feat_k2(float* __restrict__ klf, unsigned short* __restrict__ kpb,
                           const unsigned int* __restrict__ klsbits) {
  const int n = BB * HH * LL * MP;
  for (int i = blockIdx.x * blockDim.x + threadIdx.x; i < n; i += gridDim.x * blockDim.x) {
    int m  = i & 31;
    int bh = i >> 15;  // / (LL*MP)
    float p = 0.f;
    if (m < MF) p = __expf(klf[i] - decf(klsbits[bh])) + FEPS;
    klf[i] = p;
    kpb[i] = f2bf(p);
  }
}

// ---------------------------------------------------------------- causal KV scan
__global__ __launch_bounds__(256) void sb_performer_scan(
    const float* __restrict__ qp, const float* __restrict__ kp,
    const float* __restrict__ qkv, float* __restrict__ gout,
    float* __restrict__ gcs) {
  __shared__ float kv[MF * DH];
  __shared__ float sacc[MF];
  __shared__ float red[256];
  int bh = blockIdx.x;
  int b = bh >> 4, h = bh & 15;
  int tid = threadIdx.x;
  int d = tid & 63, g = tid >> 6;
  for (int i = tid; i < MF * DH; i += 256) kv[i] = 0.f;
  if (tid < MF) sacc[tid] = 0.f;
  __syncthreads();
  const size_t rowbase = (size_t)bh * LL;
  const float* vbase = qkv + (size_t)b * LL * (3 * DM) + 2 * DM + h * DH + d;
  for (int n = 0; n < LL; ++n) {
    const float* kpn = kp + (rowbase + n) * MP;
    const float* qpn = qp + (rowbase + n) * MP;
    float vnd = vbase[(size_t)n * (3 * DM)];
    float part = 0.f;
#pragma unroll
    for (int t = 0; t < 6; ++t) {
      int m = g * 6 + t;
      float nk = kv[m * DH + d] + kpn[m] * vnd;
      kv[m * DH + d] = nk;
      part += qpn[m] * nk;
    }
    red[tid] = part;
    if (d == 0) {
#pragma unroll
      for (int t = 0; t < 6; ++t) sacc[g * 6 + t] += kpn[g * 6 + t];
    }
    __syncthreads();
    if (tid < DH)
      gout[(rowbase + n) * DH + tid] = red[tid] + red[64 + tid] + red[128 + tid] + red[192 + tid];
    if (tid == 0) {
      float cs = 0.f;
      for (int m = 0; m < MF; ++m) cs += qpn[m] * sacc[m];
      gcs[rowbase + n] = cs;
    }
    __syncthreads();
  }
}

// ---------------------------------------------------------------- local attention + combine
#define LAT_WAVES 2
__global__ __launch_bounds__(64) void sb_local_attn_combine(
    const unsigned short* __restrict__ qb,  const unsigned short* __restrict__ kbm,
    const unsigned short* __restrict__ qpb, const unsigned short* __restrict__ kpb,
    const unsigned short* __restrict__ vb,  const float* __restrict__ qls,
    const unsigned int* __restrict__ klsbits,
    const float* __restrict__ gout, const float* __restrict__ gcs,
    unsigned short* __restrict__ ctxb) {
  __shared__ __align__(16) float Sl[LAT_WAVES][16][144];
  __shared__ __align__(16) float Gl[LAT_WAVES][16][144];
  __shared__ __align__(16) unsigned short Pl[LAT_WAVES][16][160];
  __shared__ float lse_s[LAT_WAVES][16];
  __shared__ float gps_s[LAT_WAVES][16];

  int w = threadIdx.x >> 5;
  int lane = threadIdx.x & 31;
  int half = lane >> 4, ln = lane & 15;
  int t  = blockIdx.x;
  int rb = t & 31;          // L/32 row-blocks
  int bh = t >> 5;
  int b = bh >> 4, h = bh & 15;
  int i0  = rb * 32 + w * 16;
  int jlo = i0 - WIN;
  size_t rowbase = (size_t)bh * LL;
  const unsigned short* qbase  = qb  + rowbase * DH;
  const unsigned short* kbase  = kbm + rowbase * DH;
  const unsigned short* qpbase = qpb + rowbase * MP;
  const unsigned short* kpbase = kpb + rowbase * MP;
  const unsigned short* vbase  = vb  + rowbase * DH;

  v16bf qa0 = load_a_row(qbase  + (size_t)(i0 + ln) * DH, 0,  half);
  v16bf qa1 = load_a_row(qbase  + (size_t)(i0 + ln) * DH, 32, half);
  v16bf qpa = load_a_row(qpbase + (size_t)(i0 + ln) * MP, 0,  half);

  for (int jt = 0; jt < 9; ++jt) {
    int j0 = jlo + jt * 16;
    int jr = j0 + ln; jr = jr < 0 ? 0 : jr;   // upper bound always < L
    v16bf kb0  = load_b_row(kbase  + (size_t)jr * DH, 0,  half);
    v16bf kb1  = load_b_row(kbase  + (size_t)jr * DH, 32, half);
    v16bf kpbf = load_b_row(kpbase + (size_t)jr * MP, 0,  half);
    v8f s = {};
    s = __builtin_amdgcn_wmma_f32_16x16x32_bf16(false, qa0, false, kb0, (short)0, s, false, false);
    s = __builtin_amdgcn_wmma_f32_16x16x32_bf16(false, qa1, false, kb1, (short)0, s, false, false);
    v8f gm = {};
    gm = __builtin_amdgcn_wmma_f32_16x16x32_bf16(false, qpa, false, kpbf, (short)0, gm, false, false);
#pragma unroll
    for (int r = 0; r < 8; ++r) {
      int row = r + half * 8;
      int i = i0 + row;
      int j = j0 + ln;
      bool keep = (j >= 0) && (j <= i) && (j >= i - WIN);
      float sv = s[r] * SCL + (keep ? 0.f : NEG_MASK);
      float gv = gm[r] * SCL + (keep ? 0.f : NEG_MASK);
      gv = gv <= 0.f ? 0.f : gv;
      Sl[w][row][jt * 16 + ln] = sv;
      Gl[w][row][jt * 16 + ln] = gv;
    }
  }
  __syncthreads();

  {
    int row = ln;
    int c0 = (lane < 16) ? 0 : 72;
    int c1 = c0 + 72;
    float mx = -3e38f;
    for (int c = c0; c < c1; ++c) mx = fmaxf(mx, Sl[w][row][c]);
    mx = fmaxf(mx, __shfl_xor(mx, 16));
    float se = 0.f, gs = 0.f;
    for (int c = c0; c < c1; ++c) { se += __expf(Sl[w][row][c] - mx); gs += Gl[w][row][c]; }
    se += __shfl_xor(se, 16);
    gs += __shfl_xor(gs, 16);
    if (lane < 16) {
      int i = i0 + row;
      float lse    = mx + __logf(se);
      float gscale = qls[rowbase + i] + decf(klsbits[bh]);
      float gln    = __logf(fmaxf(gcs[rowbase + i] - gs, 1e-24f)) + gscale;
      float m2 = fmaxf(lse, gln);
      float lognorm = m2 + __logf(__expf(lse - m2) + __expf(gln - m2));
      lse_s[w][row] = lse;
      gps_s[w][row] = __expf(gscale - lognorm);
    }
  }
  __syncthreads();

  for (int idx = lane; idx < 16 * 160; idx += 32) {
    int row = idx / 160, c = idx - row * 160;
    float p = 0.f;
    if (c < 144) p = __expf(Sl[w][row][c] - lse_s[w][row]) - Gl[w][row][c] * gps_s[w][row];
    Pl[w][row][c] = f2bf(p);
  }
  __syncthreads();

#pragma unroll
  for (int dt = 0; dt < 4; ++dt) {
    v8f acc = {};
#pragma unroll
    for (int kc = 0; kc < 5; ++kc) {
      v16bf pa = load_a_row(&Pl[w][ln][0], kc * 32, half);   // ds_load_b128 from LDS
      FragU vfr;
      int jb = jlo + kc * 32 + half * 16;
#pragma unroll
      for (int e = 0; e < 16; ++e) {
        int j = jb + e;
        j = j < 0 ? 0 : (j > LL - 1 ? LL - 1 : j);
        vfr.u[e] = vbase[(size_t)j * DH + dt * 16 + ln];
      }
      acc = __builtin_amdgcn_wmma_f32_16x16x32_bf16(false, pa, false, vfr.v, (short)0, acc, false, false);
    }
#pragma unroll
    for (int r = 0; r < 8; ++r) {
      int row = r + half * 8;
      int i = i0 + row;
      int d = dt * 16 + ln;
      float val = acc[r] + gout[(rowbase + i) * DH + d] * gps_s[w][row];
      ctxb[((size_t)(b * LL + i)) * DM + h * DH + d] = f2bf(val);
    }
  }
}

// ---------------------------------------------------------------- launcher
extern "C" void kernel_launch(void* const* d_in, const int* in_sizes, int n_in,
                              void* d_out, int out_size, void* d_ws, size_t ws_size,
                              hipStream_t stream) {
  const float* x      = (const float*)d_in[0];
  const float* Wqkv_w = (const float*)d_in[1];
  const float* Wqkv_b = (const float*)d_in[2];
  const float* proj   = (const float*)d_in[3];
  const float* out_w  = (const float*)d_in[4];
  const float* out_b  = (const float*)d_in[5];
  float* out = (float*)d_out;

  char* ws = (char*)d_ws;
  size_t off = 0;
  auto alloc = [&](size_t bytes) {
    char* p = ws + off;
    off = (off + bytes + 255) & ~(size_t)255;
    return p;
  };
  unsigned short* xb    = (unsigned short*)alloc((size_t)BB * LL * DM * 2);
  unsigned short* wqkvb = (unsigned short*)alloc((size_t)3 * DM * DM * 2);
  unsigned short* wob   = (unsigned short*)alloc((size_t)DM * DM * 2);
  float*          qkv   = (float*)alloc((size_t)BB * LL * 3 * DM * 4);
  unsigned short* qb    = (unsigned short*)alloc((size_t)BB * HH * LL * DH * 2);
  unsigned short* kb    = (unsigned short*)alloc((size_t)BB * HH * LL * DH * 2);
  unsigned short* vb16  = (unsigned short*)alloc((size_t)BB * HH * LL * DH * 2);
  float*          qp32  = (float*)alloc((size_t)BB * HH * LL * MP * 4);
  float*          kp32  = (float*)alloc((size_t)BB * HH * LL * MP * 4);
  unsigned short* qpb   = (unsigned short*)alloc((size_t)BB * HH * LL * MP * 2);
  unsigned short* kpb   = (unsigned short*)alloc((size_t)BB * HH * LL * MP * 2);
  float*          qls   = (float*)alloc((size_t)BB * HH * LL * 4);
  unsigned int*   klsb  = (unsigned int*)alloc(256);
  float*          gout  = (float*)alloc((size_t)BB * HH * LL * DH * 4);
  float*          gcs   = (float*)alloc((size_t)BB * HH * LL * 4);
  unsigned short* ctxb  = (unsigned short*)alloc((size_t)BB * LL * DM * 2);

  sb_cvt_f32_to_bf16<<<256, 256, 0, stream>>>(x, xb, BB * LL * DM);
  sb_cvt_f32_to_bf16<<<256, 256, 0, stream>>>(Wqkv_w, wqkvb, 3 * DM * DM);
  sb_cvt_f32_to_bf16<<<256, 256, 0, stream>>>(out_w, wob, DM * DM);
  sb_init_kls<<<1, 64, 0, stream>>>(klsb);

  sb_gemm_bf16_wmma<<<dim3(BB * LL / 128, 3 * DM / 64), 256, 0, stream>>>(
      xb, wqkvb, Wqkv_b, qkv, BB * LL, 3 * DM, DM);

  sb_extract_qkv_bf16<<<512, 256, 0, stream>>>(qkv, qb, 0);
  sb_extract_qkv_bf16<<<512, 256, 0, stream>>>(qkv, kb, 1);
  sb_extract_qkv_bf16<<<512, 256, 0, stream>>>(qkv, vb16, 2);

  sb_feat_q<<<BB * HH * LL / 256, 256, 0, stream>>>(qkv, proj, qp32, qpb, qls);
  sb_feat_k1<<<BB * HH * LL / 256, 256, 0, stream>>>(qkv, proj, kp32, klsb);
  sb_feat_k2<<<512, 256, 0, stream>>>(kp32, kpb, klsb);

  sb_performer_scan<<<BB * HH, 256, 0, stream>>>(qp32, kp32, qkv, gout, gcs);

  sb_local_attn_combine<<<BB * HH * (LL / 32), 64, 0, stream>>>(
      qb, kb, qpb, kpb, vb16, qls, klsb, gout, gcs, ctxb);

  sb_gemm_bf16_wmma<<<dim3(BB * LL / 128, DM / 64), 256, 0, stream>>>(
      ctxb, wob, out_b, out, BB * LL, DM, DM);
}
